// C2BFusion_29257317220742
// MI455X (gfx1250) — compile-verified, matrix operand order checked by reading, not run on hardware
//
#include <hip/hip_runtime.h>
#include <hip/hip_bf16.h>

// ---------------- CDNA5 WMMA / TDM types ----------------
typedef __attribute__((ext_vector_type(16))) __bf16 v16bf;
typedef __attribute__((ext_vector_type(8)))  float  v8f;
typedef __attribute__((ext_vector_type(4)))  unsigned v4u;
typedef __attribute__((ext_vector_type(8)))  int    v8i;
typedef __attribute__((ext_vector_type(4)))  int    v4i;

union Frag {
    uint4 u[2];
    v16bf v;
};

__device__ __forceinline__ unsigned short f2bf(float f) {
    unsigned u = __float_as_uint(f);
    u += 0x7FFFu + ((u >> 16) & 1u);   // round-to-nearest-even
    return (unsigned short)(u >> 16);
}

#if __has_builtin(__builtin_amdgcn_tensor_load_to_lds)
#define HAVE_TDM 1
#else
#define HAVE_TDM 0
#endif

// Async copy global->LDS, 16 bytes per lane (GVS mode: saddr base + vaddr off).
__device__ __forceinline__ void async_b128(unsigned lds, unsigned goff,
                                           unsigned long long base) {
    asm volatile("global_load_async_to_lds_b128 %0, %1, %2"
                 :: "v"(lds), "v"(goff), "s"(base) : "memory");
}

// Problem constants (fixed by reference setup_inputs)
#define NB      16384
#define NPTS    32768
#define CM      64
#define CO      128
#define HIMG    192
#define WIMG    640
#define YB_STRIDE 652800   // per-batch floats in y workspace (64 * 10200)

// -----------------------------------------------------------------------------
// K1: conv3x3 (Ci=256 -> Cm=64) + scale/offset + ReLU, implicit-GEMM via WMMA.
// One block = 128 output pixels x 64 output channels for one (level, batch).
// K dimension = 9 taps x 256 ci = 72 k-steps of 32 (bf16). Staging via VALU
// (padding + f32->bf16 conversion is needed, so no raw DMA here).
// -----------------------------------------------------------------------------
__global__ void conv_wmma(const float* __restrict__ x,
                          const float* __restrict__ wgt,
                          const float* __restrict__ bias,
                          const float* __restrict__ scl,
                          const float* __restrict__ offs,
                          float* __restrict__ y,
                          int h, int w, int hw) {
    const int tid  = threadIdx.x;
    const int p0   = blockIdx.x * 128;
    const int b    = blockIdx.y;
    const float* xb = x + (size_t)b * 256 * hw;

    __shared__ __align__(16) unsigned short As[128 * 40];
    __shared__ __align__(16) unsigned short Bs[64 * 40];

    const int wid  = tid >> 5;
    const int lane = tid & 31;
    const int m    = lane & 15;
    const int hi   = lane >> 4;

    v8f acc[4];
#pragma unroll
    for (int i = 0; i < 4; ++i) acc[i] = (v8f){0.f,0.f,0.f,0.f,0.f,0.f,0.f,0.f};

    for (int t = 0; t < 9; ++t) {
        const int dy = t / 3 - 1;
        const int dx = t % 3 - 1;
        for (int cb = 0; cb < 8; ++cb) {
            const int ci0 = cb * 32;
            __syncthreads();
#pragma unroll
            for (int i = 0; i < 16; ++i) {
                int idx = i * 256 + tid;
                int k   = idx >> 7;
                int pl  = idx & 127;
                int p   = p0 + pl;
                float v = 0.f;
                if (p < hw) {
                    int py = p / w;
                    int px = p - py * w;
                    int sy = py + dy, sx = px + dx;
                    if ((unsigned)sy < (unsigned)h && (unsigned)sx < (unsigned)w)
                        v = xb[(size_t)(ci0 + k) * hw + sy * w + sx];
                }
                As[pl * 40 + k] = f2bf(v);
            }
#pragma unroll
            for (int i = 0; i < 8; ++i) {
                int idx = i * 256 + tid;
                int n   = idx >> 5;
                int k   = idx & 31;
                Bs[n * 40 + k] = f2bf(wgt[(size_t)n * 2304 + (ci0 + k) * 9 + t]);
            }
            __syncthreads();

            Frag a;
            const uint4* ap = (const uint4*)(&As[(wid * 16 + m) * 40]);
            a.u[0] = ap[hi];
            a.u[1] = ap[2 + hi];
#pragma unroll
            for (int nt = 0; nt < 4; ++nt) {
                Frag bf;
                const uint4* bp = (const uint4*)(&Bs[(nt * 16 + m) * 40]);
                bf.u[0] = bp[hi];
                bf.u[1] = bp[2 + hi];
                acc[nt] = __builtin_amdgcn_wmma_f32_16x16x32_bf16(
                    false, a.v, false, bf.v, (short)0, acc[nt], false, false);
            }
        }
    }

    float* yb = y + (size_t)b * YB_STRIDE;
    const int pbase = p0 + wid * 16 + 8 * hi;
#pragma unroll
    for (int nt = 0; nt < 4; ++nt) {
        int c = nt * 16 + m;
        float bb = bias[c], ss = scl[c], oo = offs[c];
        float* yc = yb + (size_t)c * hw;
#pragma unroll
        for (int r = 0; r < 8; ++r) {
            int pp = pbase + r;
            if (pp < hw) {
                float v = (acc[nt][r] + bb) * ss + oo;
                yc[pp] = v > 0.f ? v : 0.f;
            }
        }
    }
}

// -----------------------------------------------------------------------------
// K2: pooled = separable weighted spatial mean of conv outputs (== mean of the
// bilinear-resized feats), then the level-attention MLP -> lwsc = 1+sigmoid(.)
// -----------------------------------------------------------------------------
__global__ void pooled_k(const float* __restrict__ yall,
                         const float* __restrict__ la_w1, const float* __restrict__ la_b1,
                         const float* __restrict__ la_w2, const float* __restrict__ la_b2,
                         float* __restrict__ lwsc) {
    __shared__ float Rw[90];
    __shared__ float Cw[300];
    __shared__ float pooled[512];
    __shared__ float h1[64];
    const int tid = threadIdx.x;
    const int hh[4] = {48, 24, 12, 6};
    const int wwl[4] = {160, 80, 40, 20};
    const int ro[4] = {0, 48, 72, 84};
    const int col[4] = {0, 160, 240, 280};
    const int loff[4] = {0, 491520, 614400, 645120};

    for (int i = tid; i < 90;  i += 256) Rw[i] = 0.f;
    for (int i = tid; i < 300; i += 256) Cw[i] = 0.f;
    __syncthreads();
    if (tid == 0) {
        for (int l = 0; l < 4; ++l) {
            float ky = (float)(hh[l] - 1) / 191.f;
            for (int oy = 0; oy < HIMG; ++oy) {
                float ys = oy * ky;
                int y0 = (int)ys;
                float wy = ys - (float)y0;
                int y1 = min(y0 + 1, hh[l] - 1);
                Rw[ro[l] + y0] += 1.f - wy;
                Rw[ro[l] + y1] += wy;
            }
            float kx = (float)(wwl[l] - 1) / 639.f;
            for (int ox = 0; ox < WIMG; ++ox) {
                float xs = ox * kx;
                int x0 = (int)xs;
                float wx = xs - (float)x0;
                int x1 = min(x0 + 1, wwl[l] - 1);
                Cw[col[l] + x0] += 1.f - wx;
                Cw[col[l] + x1] += wx;
            }
        }
    }
    __syncthreads();
    for (int t = tid; t < 512; t += 256) {
        int b = t >> 8, rem = t & 255, l = rem >> 6, c = rem & 63;
        const float* yb = yall + (size_t)b * YB_STRIDE + loff[l] + (size_t)c * hh[l] * wwl[l];
        float sum = 0.f;
        for (int i = 0; i < hh[l]; ++i) {
            float rs = 0.f;
            for (int j = 0; j < wwl[l]; ++j) rs += Cw[col[l] + j] * yb[i * wwl[l] + j];
            sum += Rw[ro[l] + i] * rs;
        }
        pooled[t] = sum * (1.f / (192.f * 640.f));
    }
    __syncthreads();
    if (tid < 64) {
        int b = tid >> 5, o = tid & 31;
        float s = la_b1[o];
        for (int j = 0; j < 256; ++j) s += pooled[b * 256 + j] * la_w1[o * 256 + j];
        h1[tid] = fmaxf(s, 0.f);
    }
    __syncthreads();
    if (tid < 8) {
        int b = tid >> 2, l = tid & 3;
        float s = la_b2[l];
        for (int j = 0; j < 32; ++j) s += h1[b * 32 + j] * la_w2[l * 32 + j];
        lwsc[b * 4 + l] = 1.f + 1.f / (1.f + __expf(-s));
    }
}

// Convert imgt_w (128x256 fp32) to bf16 once so the GEMM B tiles are DMA-able.
__global__ void cvtw_k(const float* __restrict__ w, unsigned short* __restrict__ o) {
    int i = blockIdx.x * 256 + threadIdx.x;
    o[i] = f2bf(w[i]);
}

// -----------------------------------------------------------------------------
// K3: projection + bilinear gather directly from conv outputs (resized image is
// never materialized). 4 threads per point, one FPN level each -> the four
// 128-byte bf16 channel chunks written by neighbor threads form one contiguous
// 512-byte gath row. Also pts_pre (32 contiguous channels per thread).
// -----------------------------------------------------------------------------
__global__ void gather_pts(const float* __restrict__ pts,
                           const float* __restrict__ calib,
                           const float* __restrict__ yall,
                           const float* __restrict__ lwsc,
                           const float* __restrict__ pw, const float* __restrict__ pb,
                           const float* __restrict__ ps, const float* __restrict__ po,
                           unsigned short* __restrict__ gath,
                           float* __restrict__ ptspre) {
    const int tid = threadIdx.x;
    const int p   = blockIdx.x * 64 + (tid >> 2);   // point index
    const int sub = tid & 3;                        // level / channel group
    const int b   = p >> 14;

    const float px = pts[p * 5 + 1];
    const float py = pts[p * 5 + 2];
    const float pz = pts[p * 5 + 3];
    const float pi = pts[p * 5 + 4];

    // pts_pre: channels [sub*32, sub*32+32)
    {
        const int c0 = sub * 32;
#pragma unroll 8
        for (int c = c0; c < c0 + 32; ++c) {
            float v = px * pw[c * 4 + 0] + py * pw[c * 4 + 1] +
                      pz * pw[c * 4 + 2] + pi * pw[c * 4 + 3];
            ptspre[(size_t)p * CO + c] = (v + pb[c]) * ps[c] + po[c];
        }
    }

    const float* C = calib + b * 12;
    float q0 = C[0] * px + C[1] * py + C[2]  * pz + C[3];
    float q1 = C[4] * px + C[5] * py + C[6]  * pz + C[7];
    float q2 = C[8] * px + C[9] * py + C[10] * pz + C[11];
    float zs = fmaxf(q2, 1e-5f);
    float u = q0 / zs, v = q1 / zs;
    bool on = (u >= 0.f) && (u < (float)WIMG) && (v >= 0.f) && (v < (float)HIMG) &&
              (q2 > 1e-5f);
    int ui = min(max((int)u, 0), WIMG - 1);
    int vi = min(max((int)v, 0), HIMG - 1);

    const int hh[4] = {48, 24, 12, 6};
    const int wwl[4] = {160, 80, 40, 20};
    const int loff[4] = {0, 491520, 614400, 645120};

    unsigned short* grow = gath + (size_t)p * 256 + sub * CM;
    if (!on) {
        for (int c = 0; c < CM; ++c) grow[c] = 0;
        return;
    }
    const int hl = hh[sub], wl = wwl[sub], hwl = hl * wl;
    float fy = (float)vi * ((float)(hl - 1) / 191.f);
    float fx = (float)ui * ((float)(wl - 1) / 639.f);
    int y0 = (int)fy;
    int x0 = (int)fx;
    float ty = fy - (float)y0;
    float tx = fx - (float)x0;
    int y1 = min(y0 + 1, hl - 1);
    int x1 = min(x0 + 1, wl - 1);
    const float s = lwsc[b * 4 + sub];
    const float* base = yall + (size_t)b * YB_STRIDE + loff[sub];
#pragma unroll 4
    for (int c = 0; c < CM; ++c) {
        const float* r0 = base + (size_t)c * hwl + y0 * wl;
        const float* r1 = base + (size_t)c * hwl + y1 * wl;
        float top = r0[x0] * (1.f - tx) + r0[x1] * tx;
        float bot = r1[x0] * (1.f - tx) + r1[x1] * tx;
        grow[c] = f2bf((top * (1.f - ty) + bot * ty) * s);
    }
}

// -----------------------------------------------------------------------------
// K4: img_pre = (gath[N,256] @ imgt_w^T + b)*s + o via bf16 WMMA.
// Block = 128 points x 128 channels; double-buffered LDS:
//   A tile via GLOBAL_LOAD_ASYNC_TO_LDS_B128 (ASYNCcnt),
//   B tile via Tensor Data Mover (TENSORcnt) when available.
// -----------------------------------------------------------------------------
__global__ void imggemm_wmma(const unsigned short* __restrict__ gth,
                             const unsigned short* __restrict__ wbf,  // bf16 (128,256)
                             const float* __restrict__ bias,
                             const float* __restrict__ scl,
                             const float* __restrict__ offs,
                             float* __restrict__ outp) {
    const int tid  = threadIdx.x;
    const int p0   = blockIdx.x * 128;
    const int wid  = tid >> 5;
    const int lane = tid & 31;
    const int m    = lane & 15;
    const int hi   = lane >> 4;
    const int mg   = wid >> 2;
    const int ng   = wid & 3;

    __shared__ __align__(16) unsigned short As[2][128 * 40];
    __shared__ __align__(16) unsigned short Bs[2][128 * 40];

    const unsigned asA[2] = {(unsigned)(size_t)&As[0][0], (unsigned)(size_t)&As[1][0]};
    const unsigned asB[2] = {(unsigned)(size_t)&Bs[0][0], (unsigned)(size_t)&Bs[1][0]};
    const unsigned long long gbase = (unsigned long long)(size_t)gth;
    const unsigned long long wbase = (unsigned long long)(size_t)wbf;

    auto issue = [&](int kk, int d) {
        const int ci0 = kk * 32;
        // A tile: 128 rows x 64B, 2 async b128 per thread
#pragma unroll
        for (int i = 0; i < 2; ++i) {
            int idx = i * 256 + tid;
            int row = idx >> 2, seg = idx & 3;
            unsigned lds = asA[d] + (unsigned)((row * 40 + seg * 8) * 2);
            unsigned go  = (unsigned)((p0 + row) * 512 + ci0 * 2 + seg * 16);
            async_b128(lds, go, gbase);
        }
#if HAVE_TDM
        // B tile: one TDM 2D descriptor per wave-0: 128 rows x 64B, row stride
        // 512B, data_size=8B, LDS pad 4 DW every 16 DW -> 80B padded rows.
        if (tid < 32) {
            unsigned long long ga = wbase + (unsigned)(ci0 * 2);
            v4u g0;
            g0[0] = 1u;                                  // count=1
            g0[1] = asB[d];                              // lds_addr
            g0[2] = (unsigned)ga;                        // global_addr[31:0]
            g0[3] = (unsigned)(ga >> 32) | (2u << 30);   // addr[56:32] | type=2
            v8i g1;
            g1[0] = (int)((3u << 16) |                   // data_size = 8B
                          (1u << 20) |                   // pad_enable
                          (3u << 22) |                   // pad_interval: 16 DW
                          (3u << 25));                   // pad_amount: 4 DW
            g1[1] = (int)(64u << 16);                    // tensor_dim0 = 64 (x8B)
            g1[2] = (int)(128u << 16);                   // tensor_dim1 = 128
            g1[3] = (int)(8u << 16);                     // tile_dim0 = 8 (x8B)
            g1[4] = (int)128u;                           // tile_dim1 = 128
            g1[5] = 64;                                  // dim0 stride = 64 (x8B)
            g1[6] = 0;
            g1[7] = 0;
            v4i g2 = {0, 0, 0, 0}, g3 = {0, 0, 0, 0};
#if defined(__clang_major__) && __clang_major__ >= 23
            v8i gX = {0, 0, 0, 0, 0, 0, 0, 0};
            __builtin_amdgcn_tensor_load_to_lds(g0, g1, g2, g3, gX, 0);
#else
            __builtin_amdgcn_tensor_load_to_lds(g0, g1, g2, g3, 0);
#endif
        }
#else
        // B tile via async copies as well
#pragma unroll
        for (int i = 0; i < 2; ++i) {
            int idx = i * 256 + tid;
            int row = idx >> 2, seg = idx & 3;
            unsigned lds = asB[d] + (unsigned)((row * 40 + seg * 8) * 2);
            unsigned go  = (unsigned)(row * 512 + ci0 * 2 + seg * 16);
            async_b128(lds, go, wbase);
        }
#endif
    };

    v8f acc[4][2];
#pragma unroll
    for (int i = 0; i < 4; ++i)
#pragma unroll
        for (int j = 0; j < 2; ++j)
            acc[i][j] = (v8f){0.f,0.f,0.f,0.f,0.f,0.f,0.f,0.f};

    issue(0, 0);

    for (int kk = 0; kk < 8; ++kk) {
        const int buf = kk & 1;
        if (kk < 7) {
            issue(kk + 1, buf ^ 1);
#if HAVE_TDM
            asm volatile("s_wait_asynccnt 2" ::: "memory");
            if (tid < 32) __builtin_amdgcn_s_wait_tensorcnt(1);
#else
            asm volatile("s_wait_asynccnt 4" ::: "memory");
#endif
        } else {
            asm volatile("s_wait_asynccnt 0" ::: "memory");
#if HAVE_TDM
            if (tid < 32) __builtin_amdgcn_s_wait_tensorcnt(0);
#endif
        }
        __syncthreads();

        const unsigned short* Ab = &As[buf][0];
        const unsigned short* Bb = &Bs[buf][0];
        Frag a[4], bf[2];
#pragma unroll
        for (int ms = 0; ms < 4; ++ms) {
            const uint4* ap = (const uint4*)(Ab + (mg * 64 + ms * 16 + m) * 40);
            a[ms].u[0] = ap[hi];
            a[ms].u[1] = ap[2 + hi];
        }
#pragma unroll
        for (int ns = 0; ns < 2; ++ns) {
            const uint4* bp = (const uint4*)(Bb + ((ng * 2 + ns) * 16 + m) * 40);
            bf[ns].u[0] = bp[hi];
            bf[ns].u[1] = bp[2 + hi];
        }
#pragma unroll
        for (int ms = 0; ms < 4; ++ms)
#pragma unroll
            for (int ns = 0; ns < 2; ++ns)
                acc[ms][ns] = __builtin_amdgcn_wmma_f32_16x16x32_bf16(
                    false, a[ms].v, false, bf[ns].v, (short)0, acc[ms][ns], false, false);
        __syncthreads();   // protect buf from issue(kk+2) overwrite
    }

#pragma unroll
    for (int ns = 0; ns < 2; ++ns) {
        int c = (ng * 2 + ns) * 16 + m;
        float bb = bias[c], ss = scl[c], oo = offs[c];
#pragma unroll
        for (int ms = 0; ms < 4; ++ms) {
            int ptb = p0 + mg * 64 + ms * 16 + 8 * hi;
#pragma unroll
            for (int r = 0; r < 8; ++r)
                outp[(size_t)(ptb + r) * CO + c] = (acc[ms][ns][r] + bb) * ss + oo;
        }
    }
}

// -----------------------------------------------------------------------------
// K5a: channel-wise means over all N points (cross = [img_pool, pts_pool])
// -----------------------------------------------------------------------------
__global__ void pool_reduce(const float* __restrict__ imgpre,
                            const float* __restrict__ ptspre,
                            float* __restrict__ pools) {
    const int c = blockIdx.x;
    const float* src = (c < CO) ? imgpre : ptspre;
    const int ch = c & (CO - 1);
    float s = 0.f;
    for (int i = threadIdx.x; i < NPTS; i += 256)
        s += src[(size_t)i * CO + ch];
    __shared__ float red[256];
    red[threadIdx.x] = s;
    __syncthreads();
    for (int st = 128; st > 0; st >>= 1) {
        if (threadIdx.x < st) red[threadIdx.x] += red[threadIdx.x + st];
        __syncthreads();
    }
    if (threadIdx.x == 0) pools[c] = red[0] * (1.f / (float)NPTS);
}

// K5b: SE gates. gates[0..127]=1+g_pt, [128..255]=1+g_img.
__global__ void gates_k(const float* __restrict__ pools,
                        const float* __restrict__ w1p, const float* __restrict__ w2p,
                        const float* __restrict__ w1i, const float* __restrict__ w2i,
                        float* __restrict__ gates) {
    __shared__ float cr[256];
    __shared__ float h[256];
    const int tid = threadIdx.x;
    cr[tid] = pools[tid];
    __syncthreads();
    {
        const float* w1 = (tid < 128) ? w1p : w1i;
        int r = tid & 127;
        float s = 0.f;
        for (int j = 0; j < 256; ++j) s += cr[j] * w1[r * 256 + j];
        h[tid] = fmaxf(s, 0.f);
    }
    __syncthreads();
    {
        const float* w2 = (tid < 128) ? w2p : w2i;
        const float* hb = (tid < 128) ? h : (h + 128);
        int r = tid & 127;
        float s = 0.f;
        for (int j = 0; j < 128; ++j) s += hb[j] * w2[r * 128 + j];
        gates[tid] = 1.f + 1.f / (1.f + __expf(-s));
    }
}

// K6: fuse = relu(img_pre*(1+g_pt) + pts_pre*(1+g_img))
__global__ void fuse_k(const float* __restrict__ imgpre,
                       const float* __restrict__ ptspre,
                       const float* __restrict__ gates,
                       float* __restrict__ out) {
    const int idx = blockIdx.x * 256 + threadIdx.x;
    const int c = idx & (CO - 1);
    float v = imgpre[idx] * gates[c] + ptspre[idx] * gates[CO + c];
    out[idx] = fmaxf(v, 0.f);
}

// -----------------------------------------------------------------------------
static inline size_t al256(size_t x) { return (x + 255) & ~(size_t)255; }

extern "C" void kernel_launch(void* const* d_in, const int* in_sizes, int n_in,
                              void* d_out, int out_size, void* d_ws, size_t ws_size,
                              hipStream_t stream) {
    const float* fpn[4]  = {(const float*)d_in[0], (const float*)d_in[1],
                            (const float*)d_in[2], (const float*)d_in[3]};
    const float* points  = (const float*)d_in[4];
    const float* calib   = (const float*)d_in[5];
    const float* lat_w   = (const float*)d_in[6];
    const float* lat_b   = (const float*)d_in[7];
    const float* lat_s   = (const float*)d_in[8];
    const float* lat_o   = (const float*)d_in[9];
    const float* pts_w   = (const float*)d_in[10];
    const float* pts_b   = (const float*)d_in[11];
    const float* pts_s   = (const float*)d_in[12];
    const float* pts_o   = (const float*)d_in[13];
    const float* imgt_w  = (const float*)d_in[14];
    const float* imgt_b  = (const float*)d_in[15];
    const float* imgt_s  = (const float*)d_in[16];
    const float* imgt_o  = (const float*)d_in[17];
    const float* la_w1   = (const float*)d_in[18];
    const float* la_b1   = (const float*)d_in[19];
    const float* la_w2   = (const float*)d_in[20];
    const float* la_b2   = (const float*)d_in[21];
    const float* se_pt_w1  = (const float*)d_in[22];
    const float* se_pt_w2  = (const float*)d_in[23];
    const float* se_img_w1 = (const float*)d_in[24];
    const float* se_img_w2 = (const float*)d_in[25];

    char* ws = (char*)d_ws;
    size_t o = 0;
    float* y_ws = (float*)(ws + o);                   o += al256((size_t)2 * YB_STRIDE * 4);
    float* lwsc = (float*)(ws + o);                   o += al256(8 * 4);
    unsigned short* gath = (unsigned short*)(ws + o); o += al256((size_t)NPTS * 256 * 2);
    float* ptspre = (float*)(ws + o);                 o += al256((size_t)NPTS * CO * 4);
    float* imgpre = (float*)(ws + o);                 o += al256((size_t)NPTS * CO * 4);
    float* pools = (float*)(ws + o);                  o += al256(256 * 4);
    float* gates = (float*)(ws + o);                  o += al256(256 * 4);
    unsigned short* wbf = (unsigned short*)(ws + o);  o += al256((size_t)CO * 256 * 2);

    const int hh[4]   = {48, 24, 12, 6};
    const int wwl[4]  = {160, 80, 40, 20};
    const int loff[4] = {0, 491520, 614400, 645120};

    cvtw_k<<<(CO * 256) / 256, 256, 0, stream>>>(imgt_w, wbf);
    for (int l = 0; l < 4; ++l) {
        int hw = hh[l] * wwl[l];
        dim3 grid((hw + 127) / 128, 2);
        conv_wmma<<<grid, 256, 0, stream>>>(fpn[l], lat_w + (size_t)l * 147456,
                                            lat_b + l * 64, lat_s + l * 64, lat_o + l * 64,
                                            y_ws + loff[l], hh[l], wwl[l], hw);
    }
    pooled_k<<<1, 256, 0, stream>>>(y_ws, la_w1, la_b1, la_w2, la_b2, lwsc);
    gather_pts<<<NPTS / 64, 256, 0, stream>>>(points, calib, y_ws, lwsc,
                                              pts_w, pts_b, pts_s, pts_o, gath, ptspre);
    imggemm_wmma<<<NPTS / 128, 256, 0, stream>>>(gath, wbf, imgt_b, imgt_s, imgt_o, imgpre);
    pool_reduce<<<256, 256, 0, stream>>>(imgpre, ptspre, pools);
    gates_k<<<1, 256, 0, stream>>>(pools, se_pt_w1, se_pt_w2, se_img_w1, se_img_w2, gates);
    fuse_k<<<(NPTS * CO) / 256, 256, 0, stream>>>(imgpre, ptspre, gates, (float*)d_out);
}
